// BiasedAttentionTransformer_31945966748231
// MI455X (gfx1250) — compile-verified, matrix operand order checked by reading, not run on hardware
//
#include <hip/hip_runtime.h>

// ---------------------------------------------------------------------------
// BiasedAttentionTransformer forward for MI455X (gfx1250, wave32, WMMA).
// GEMMs: v_wmma_f32_16x16x32_bf16, weights pre-converted to transposed bf16,
// W tiles staged into LDS by the Tensor Data Mover (double buffered).
// ---------------------------------------------------------------------------

typedef __bf16 bf16;
typedef __attribute__((ext_vector_type(16))) __bf16 v16bf;
typedef __attribute__((ext_vector_type(8)))  __bf16 v8bf;
typedef __attribute__((ext_vector_type(4)))  __bf16 v4bf;
typedef __attribute__((ext_vector_type(8)))  float  v8f;
typedef __attribute__((ext_vector_type(4)))  float  v4f;
typedef __attribute__((ext_vector_type(4)))  unsigned int u32x4;
typedef __attribute__((ext_vector_type(8)))  int   i32x8;
typedef __attribute__((ext_vector_type(4)))  int   i32x4;

#define B_  32
#define L_  128
#define E_  768
#define H_  12
#define DH_ 64
#define FF_ 3072
#define D_  8
#define VP_ 136   // padded row length for transposed V tile (16B-aligned rows)

#if __has_builtin(__builtin_amdgcn_tensor_load_to_lds)
#define USE_TDM 1
#endif

// Build a 16-element bf16 A/B fragment from two contiguous 16-byte LDS runs.
__device__ __forceinline__ v16bf frag_from_lds(const bf16* p) {
  v8bf lo = *(const v8bf*)p;
  v8bf hi = *(const v8bf*)(p + 16);
  return __builtin_shufflevector(lo, hi,
      0,1,2,3,4,5,6,7,8,9,10,11,12,13,14,15);
}
// Same from global fp32 (two 32-byte runs), converting to bf16.
__device__ __forceinline__ v16bf frag_from_f32(const float* p) {
  v8f lo = *(const v8f*)p;
  v8f hi = *(const v8f*)(p + 16);
  return __builtin_shufflevector(__builtin_convertvector(lo, v8bf),
                                 __builtin_convertvector(hi, v8bf),
      0,1,2,3,4,5,6,7,8,9,10,11,12,13,14,15);
}

// ---------------------------------------------------------------------------
// Transpose + convert: Wt[n][k] = (bf16) W[k][n].  Grid (N/32, K/32), 256 thr.
// ---------------------------------------------------------------------------
__global__ __launch_bounds__(256) void convT_kernel(
    const float* __restrict__ W, bf16* __restrict__ Wt, int K, int N)
{
  __shared__ float tile[32][33];
  int n0 = blockIdx.x * 32, k0 = blockIdx.y * 32;
  int tx = threadIdx.x & 31, ty = threadIdx.x >> 5;   // 32 x 8
  for (int i = ty; i < 32; i += 8)
    tile[i][tx] = W[(size_t)(k0 + i) * N + n0 + tx];
  __syncthreads();
  for (int i = ty; i < 32; i += 8)
    Wt[(size_t)(n0 + i) * K + k0 + tx] = (bf16)tile[tx][i];
}

// ---------------------------------------------------------------------------
__global__ __launch_bounds__(256) void embed_kernel(
    const int* __restrict__ tokens, const float* __restrict__ embed,
    float* __restrict__ e)
{
  int row = blockIdx.x;
  int tok = tokens[row];
  const float* src = embed + (size_t)tok * E_;
  float* dst = e + (size_t)row * E_;
  for (int c = threadIdx.x; c < E_; c += 256) dst[c] = src[c];
}

// ---------------------------------------------------------------------------
__global__ __launch_bounds__(128) void dist_kernel(
    const float* __restrict__ r, float* __restrict__ dmat)
{
  int i = blockIdx.x, b = blockIdx.y;
  const float* ri = r + ((size_t)b * L_ + i) * 3;
  float x = ri[0], y = ri[1], z = ri[2];
  int j = threadIdx.x;
  const float* rj = r + ((size_t)b * L_ + j) * 3;
  float dx = x - rj[0], dy = y - rj[1], dz = z - rj[2];
  dmat[((size_t)b * L_ + i) * L_ + j] = sqrtf(dx*dx + dy*dy + dz*dz);
}

// ---------------------------------------------------------------------------
// LayerNorm (E=768) with optional fused residual add.
// ---------------------------------------------------------------------------
__global__ __launch_bounds__(256) void ln_kernel(
    const float* __restrict__ x, const float* __restrict__ xadd,
    const float* __restrict__ g, const float* __restrict__ bta,
    float* __restrict__ out)
{
  __shared__ float red[16];
  int row = blockIdx.x;
  int tid = threadIdx.x, lane = tid & 31, wave = tid >> 5;
  const float* xr = x + (size_t)row * E_;
  const float* ar = xadd ? xadd + (size_t)row * E_ : nullptr;

  float vals[3];
  float s = 0.f;
#pragma unroll
  for (int i = 0; i < 3; ++i) {
    float t = xr[tid + i * 256];
    if (ar) t += ar[tid + i * 256];
    vals[i] = t; s += t;
  }
  for (int off = 16; off; off >>= 1) s += __shfl_xor(s, off, 32);
  if (lane == 0) red[wave] = s;
  __syncthreads();
  float mean;
  {
    float t = 0.f;
#pragma unroll
    for (int w = 0; w < 8; ++w) t += red[w];
    mean = t * (1.0f / E_);
  }
  float vs = 0.f;
#pragma unroll
  for (int i = 0; i < 3; ++i) { float d = vals[i] - mean; vs += d * d; }
  for (int off = 16; off; off >>= 1) vs += __shfl_xor(vs, off, 32);
  if (lane == 0) red[8 + wave] = vs;
  __syncthreads();
  float var;
  {
    float t = 0.f;
#pragma unroll
    for (int w = 0; w < 8; ++w) t += red[8 + w];
    var = t * (1.0f / E_);
  }
  float inv = rsqrtf(var + 1e-5f);
  float* orow = out + (size_t)row * E_;
#pragma unroll
  for (int i = 0; i < 3; ++i) {
    int c = tid + i * 256;
    orow[c] = (vals[i] - mean) * inv * g[c] + bta[c];
  }
}

// ---------------------------------------------------------------------------
// WMMA GEMM: C[M,N] = act(A[M,K] @ W + bias[N]).   W given TRANSPOSED bf16:
// Wt[n][k].  256 thr = 8 waves; block tile 256(M) x 64(N); K-step 32.
// Each wave owns 32 rows (2 A-frags) so every B fragment feeds 2 WMMAs.
// W tile (64x32 bf16 = 4 KB) staged by TDM (tensor_load_to_lds), double
// buffered: issue DMA for next tile, compute current, then s_wait_tensorcnt.
// ---------------------------------------------------------------------------
__global__ __launch_bounds__(256) void gemm_bias_kernel(
    const float* __restrict__ A, const bf16* __restrict__ Wt,
    const float* __restrict__ bias, float* __restrict__ C,
    int M, int N, int K, int relu)
{
  __shared__ bf16 sBT[2][64 * 32];   // [n][k] tiles, 2 x 4 KB
  const int tid = threadIdx.x, lane = tid & 31, wave = tid >> 5;
  const int bm = blockIdx.y * 256, bn = blockIdx.x * 64;
  const int m = lane & 15, klo = (lane < 16) ? 0 : 8;

  v8f acc[2][4] = {};

#ifdef USE_TDM
  auto stageB = [&](int buf, int k0) {
    if (wave == 0) {
      unsigned lds = (unsigned)(size_t)(&sBT[buf][0]);
      unsigned long long ga =
          (unsigned long long)(size_t)(Wt + (size_t)bn * K + k0);
      u32x4 g0 = { 1u,                                  // count=1, no gather
                   lds,                                 // lds_addr (bytes)
                   (unsigned)(ga & 0xFFFFFFFFu),
                   (unsigned)((ga >> 32) & 0x1FFFFFFu) | (2u << 30) }; // type=2
      i32x8 g1 = { (int)(1u << 16),                     // data_size = 2B
                   (int)(((unsigned)K & 0xFFFFu) << 16),        // dim0 lo16
                   (int)(((unsigned)K >> 16) | (((unsigned)N & 0xFFFFu) << 16)),
                   (int)(((unsigned)N >> 16) | (32u << 16)),    // tile_dim0=32
                   64,                                  // tile_dim1=64
                   (int)(unsigned)K,                    // dim0_stride = K
                   0, 0 };
      i32x4 z4 = { 0, 0, 0, 0 };
      i32x8 z8 = { 0, 0, 0, 0, 0, 0, 0, 0 };
      __builtin_amdgcn_tensor_load_to_lds(g0, g1, z4, z4, z8, 0);
    }
  };
  auto waitB = [&]() {
    if (wave == 0) __builtin_amdgcn_s_wait_tensorcnt(0);
    __syncthreads();
  };
#else
  auto stageB = [&](int buf, int k0) {
    int n = tid >> 2, koff = (tid & 3) * 8;             // 256 x 16B = 4 KB
    *(v8bf*)&sBT[buf][n * 32 + koff] =
        *(const v8bf*)&Wt[(size_t)(bn + n) * K + k0 + koff];
  };
  auto waitB = [&]() { __syncthreads(); };
#endif

  stageB(0, 0);
  waitB();

  const int nk = K / 32;
  const float* arow0 = A + (size_t)(bm + wave * 32 + m) * K + klo;
  const float* arow1 = arow0 + (size_t)16 * K;

  for (int kt = 0; kt < nk; ++kt) {
    int cur = kt & 1;
    if (kt + 1 < nk) stageB(cur ^ 1, (kt + 1) * 32);

    v16bf af0 = frag_from_f32(arow0 + kt * 32);
    v16bf af1 = frag_from_f32(arow1 + kt * 32);
    v16bf b0 = frag_from_lds(&sBT[cur][(0 * 16 + m) * 32 + klo]);
    v16bf b1 = frag_from_lds(&sBT[cur][(1 * 16 + m) * 32 + klo]);
    v16bf b2 = frag_from_lds(&sBT[cur][(2 * 16 + m) * 32 + klo]);
    v16bf b3 = frag_from_lds(&sBT[cur][(3 * 16 + m) * 32 + klo]);

    acc[0][0] = __builtin_amdgcn_wmma_f32_16x16x32_bf16(false, af0, false, b0, (short)0, acc[0][0], false, false);
    acc[1][0] = __builtin_amdgcn_wmma_f32_16x16x32_bf16(false, af1, false, b0, (short)0, acc[1][0], false, false);
    acc[0][1] = __builtin_amdgcn_wmma_f32_16x16x32_bf16(false, af0, false, b1, (short)0, acc[0][1], false, false);
    acc[1][1] = __builtin_amdgcn_wmma_f32_16x16x32_bf16(false, af1, false, b1, (short)0, acc[1][1], false, false);
    acc[0][2] = __builtin_amdgcn_wmma_f32_16x16x32_bf16(false, af0, false, b2, (short)0, acc[0][2], false, false);
    acc[1][2] = __builtin_amdgcn_wmma_f32_16x16x32_bf16(false, af1, false, b2, (short)0, acc[1][2], false, false);
    acc[0][3] = __builtin_amdgcn_wmma_f32_16x16x32_bf16(false, af0, false, b3, (short)0, acc[0][3], false, false);
    acc[1][3] = __builtin_amdgcn_wmma_f32_16x16x32_bf16(false, af1, false, b3, (short)0, acc[1][3], false, false);

    waitB();
  }

  const int mh = (lane >> 4) * 8;
#pragma unroll
  for (int rg = 0; rg < 2; ++rg) {
#pragma unroll
    for (int tj = 0; tj < 4; ++tj) {
      int col = bn + tj * 16 + m;
      float bv = bias ? bias[col] : 0.f;
#pragma unroll
      for (int g = 0; g < 8; ++g) {
        int row = bm + wave * 32 + rg * 16 + mh + g;
        float v = acc[rg][tj][g] + bv;
        if (relu) v = fmaxf(v, 0.f);
        C[(size_t)row * N + col] = v;
      }
    }
  }
}

// ---------------------------------------------------------------------------
// Attention: one workgroup per (b,h). LDS = 50 KB.
// ---------------------------------------------------------------------------
__global__ __launch_bounds__(256) void attn_kernel(
    const float* __restrict__ q, const float* __restrict__ k,
    const float* __restrict__ v, const float* __restrict__ dmat,
    const unsigned char* __restrict__ padding,
    const float* __restrict__ mu, const float* __restrict__ sigma,
    const float* __restrict__ scalep, int layer, float* __restrict__ o)
{
  __shared__ bf16 sS[L_ * L_];       // 32 KB
  __shared__ bf16 sKV[DH_ * VP_];    // K tile first, then V^T (padded rows)

  const int tid = threadIdx.x, lane = tid & 31, wave = tid >> 5;
  const int b = blockIdx.x / H_, h = blockIdx.x % H_;
  const float mu_p = mu[layer * H_ + h];
  const float sg_p = sigma[layer * H_ + h];
  const float sc_p = scalep[layer * H_ + h];

  const int m    = lane & 15;
  const int klo  = (lane < 16) ? 0 : 8;
  const int mh   = (lane >> 4) * 8;
  const int irow = wave * 16 + m;

  // stage K row-major (j x 64)
  for (int p = tid; p < L_ * DH_ / 2; p += 256) {
    int dh = (p & 31) * 2, j = p >> 5;
    const float* kg = k + ((size_t)(b * L_ + j)) * E_ + h * DH_ + dh;
    union { bf16 hh[2]; unsigned u; } pk;
    pk.hh[0] = (bf16)kg[0]; pk.hh[1] = (bf16)kg[1];
    *(unsigned*)&sKV[j * DH_ + dh] = pk.u;
  }
  __syncthreads();

  // S = Q K^T
  v8f acc[8] = {};
  const float* qrow = q + ((size_t)(b * L_ + irow)) * E_ + h * DH_ + klo;
  for (int ks = 0; ks < DH_; ks += 32) {
    v16bf af = frag_from_f32(qrow + ks);
#pragma unroll
    for (int tj = 0; tj < 8; ++tj) {
      v16bf bfrag = frag_from_lds(&sKV[(tj * 16 + m) * DH_ + ks + klo]);
      acc[tj] = __builtin_amdgcn_wmma_f32_16x16x32_bf16(
          false, af, false, bfrag, (short)0, acc[tj], false, false);
    }
  }
#pragma unroll
  for (int tj = 0; tj < 8; ++tj) {
    int j = tj * 16 + m;
    bool pj = padding[b * L_ + j] != 0;
#pragma unroll
    for (int g = 0; g < 8; ++g) {
      int i = wave * 16 + mh + g;
      float dd = dmat[((size_t)(b * L_ + i)) * L_ + j];
      float t  = (dd - mu_p) / sg_p;
      float s  = acc[tj][g] * 0.125f + sc_p * __expf(-t * t);
      bool pi  = padding[b * L_ + i] != 0;
      if (i == j || pi || pj) s = -1.0e9f;
      sS[i * L_ + j] = (bf16)s;
    }
  }
  __syncthreads();   // K reads done, S complete -> sKV reusable

  // restage V transposed: sKV[dh][j], rows padded to 136 elems
  for (int p = tid; p < L_ * DH_ / 2; p += 256) {
    int dh = p & 63, j = (p >> 6) * 2;
    const float* vg = v + ((size_t)(b * L_ + j)) * E_ + h * DH_ + dh;
    union { bf16 hh[2]; unsigned u; } pk;
    pk.hh[0] = (bf16)vg[0];
    pk.hh[1] = (bf16)vg[E_];
    *(unsigned*)&sKV[dh * VP_ + j] = pk.u;
  }

  // softmax
  for (int rr = 0; rr < 16; ++rr) {
    int i = wave * 16 + rr;
    v4bf pv = *(const v4bf*)&sS[i * L_ + lane * 4];
    v4f vals = __builtin_convertvector(pv, v4f);
    float mx = fmaxf(fmaxf(vals.x, vals.y), fmaxf(vals.z, vals.w));
    for (int off = 16; off; off >>= 1) mx = fmaxf(mx, __shfl_xor(mx, off, 32));
    vals.x = __expf(vals.x - mx); vals.y = __expf(vals.y - mx);
    vals.z = __expf(vals.z - mx); vals.w = __expf(vals.w - mx);
    float sum = vals.x + vals.y + vals.z + vals.w;
    for (int off = 16; off; off >>= 1) sum += __shfl_xor(sum, off, 32);
    float inv = 1.0f / sum;
    vals *= inv;
    *(v4bf*)&sS[i * L_ + lane * 4] = __builtin_convertvector(vals, v4bf);
  }
  __syncthreads();

  // O = P @ V
  v8f oacc[4] = {};
  for (int ks = 0; ks < L_; ks += 32) {
    v16bf af = frag_from_lds(&sS[irow * L_ + ks + klo]);
#pragma unroll
    for (int tn = 0; tn < 4; ++tn) {
      v16bf bfrag = frag_from_lds(&sKV[(tn * 16 + m) * VP_ + ks + klo]);
      oacc[tn] = __builtin_amdgcn_wmma_f32_16x16x32_bf16(
          false, af, false, bfrag, (short)0, oacc[tn], false, false);
    }
  }
#pragma unroll
  for (int tn = 0; tn < 4; ++tn) {
    int dh = tn * 16 + m;
#pragma unroll
    for (int g = 0; g < 8; ++g) {
      int i = wave * 16 + mh + g;
      o[((size_t)(b * L_ + i)) * E_ + h * DH_ + dh] = oacc[tn][g];
    }
  }
}

// ---------------------------------------------------------------------------
__global__ __launch_bounds__(256) void addmask_kernel(
    const float* __restrict__ e3, const float* __restrict__ e2,
    const unsigned char* __restrict__ pad, float* __restrict__ e)
{
  size_t idx = (size_t)blockIdx.x * 256 + threadIdx.x;
  int row = (int)(idx / E_);
  float t = pad[row] ? 0.f : e3[idx];
  e[idx] = t + e2[idx];
}

// ---------------------------------------------------------------------------
__global__ __launch_bounds__(256) void pool_kernel(
    const float* __restrict__ eF, float* __restrict__ pooled)
{
  int b = blockIdx.y;
  int c = blockIdx.x * 256 + threadIdx.x;
  float s = 0.f;
  for (int l = 0; l < L_; ++l) s += eF[((size_t)(b * L_ + l)) * E_ + c];
  pooled[b * E_ + c] = s * (1.0f / L_);
}

// ---------------------------------------------------------------------------
__global__ __launch_bounds__(256) void out_kernel(
    const float* __restrict__ pooled, const float* __restrict__ Wout,
    const float* __restrict__ bout, float* __restrict__ out)
{
  __shared__ float red[8];
  int b = blockIdx.x, tid = threadIdx.x;
  float s = 0.f;
  for (int c = tid; c < E_; c += 256) s += pooled[b * E_ + c] * Wout[c];
  for (int off = 16; off; off >>= 1) s += __shfl_xor(s, off, 32);
  if ((tid & 31) == 0) red[tid >> 5] = s;
  __syncthreads();
  if (tid == 0) {
    float t = 0.f;
#pragma unroll
    for (int w = 0; w < 8; ++w) t += red[w];
    out[b] = t + bout[0];
  }
}

// ---------------------------------------------------------------------------
extern "C" void kernel_launch(void* const* d_in, const int* in_sizes, int n_in,
                              void* d_out, int out_size, void* d_ws, size_t ws_size,
                              hipStream_t stream) {
  const int*           tokens  = (const int*)d_in[0];
  const unsigned char* padding = (const unsigned char*)d_in[1];
  const float* r      = (const float*)d_in[2];
  const float* embed  = (const float*)d_in[3];
  const float* mu     = (const float*)d_in[4];
  const float* sigma  = (const float*)d_in[5];
  const float* scalep = (const float*)d_in[6];
  const float* Wq = (const float*)d_in[7];
  const float* bq = (const float*)d_in[8];
  const float* Wk = (const float*)d_in[9];
  const float* bk = (const float*)d_in[10];
  const float* Wv = (const float*)d_in[11];
  const float* bv = (const float*)d_in[12];
  const float* Wo = (const float*)d_in[13];
  const float* bo = (const float*)d_in[14];
  const float* ln1_g = (const float*)d_in[15];
  const float* ln1_b = (const float*)d_in[16];
  const float* ln2_g = (const float*)d_in[17];
  const float* ln2_b = (const float*)d_in[18];
  const float* W1 = (const float*)d_in[19];
  const float* b1 = (const float*)d_in[20];
  const float* W2 = (const float*)d_in[21];
  const float* b2 = (const float*)d_in[22];
  const float* lnf_g = (const float*)d_in[23];
  const float* lnf_b = (const float*)d_in[24];
  const float* Wout  = (const float*)d_in[25];
  const float* bout  = (const float*)d_in[26];

  const size_t SZ_E = (size_t)B_ * L_ * E_;
  const size_t SZ_H = (size_t)B_ * L_ * FF_;
  const size_t SZ_D = (size_t)B_ * L_ * L_;

  float* w = (float*)d_ws;
  size_t off = 0;
  float* dmat  = w + off; off += SZ_D;
  float* e     = w + off; off += SZ_E;
  float* e0    = w + off; off += SZ_E;
  float* qb    = w + off; off += SZ_E;
  float* kb    = w + off; off += SZ_E;
  float* vb    = w + off; off += SZ_E;
  float* ob    = w + off; off += SZ_E;
  float* e1    = w + off; off += SZ_E;
  float* e2    = w + off; off += SZ_E;
  float* e3    = w + off; off += SZ_E;
  float* hbuf  = w + off; off += SZ_H;
  float* pooled= w + off; off += (size_t)B_ * E_;

  // bf16 transposed weights in the remaining workspace
  bf16* wt = (bf16*)(w + off);
  size_t boff = 0;
  bf16* wtq = wt + boff; boff += (size_t)D_ * E_ * E_;
  bf16* wtk = wt + boff; boff += (size_t)D_ * E_ * E_;
  bf16* wtv = wt + boff; boff += (size_t)D_ * E_ * E_;
  bf16* wto = wt + boff; boff += (size_t)D_ * E_ * E_;
  bf16* wt1 = wt + boff; boff += (size_t)D_ * E_ * FF_;
  bf16* wt2 = wt + boff; boff += (size_t)D_ * FF_ * E_;

  // ---- weight transpose/convert (once per call; trivial bandwidth)
  dim3 tEE(E_ / 32, E_ / 32);          // K=E, N=E
  dim3 tEF(FF_ / 32, E_ / 32);         // K=E, N=FF
  dim3 tFE(E_ / 32, FF_ / 32);         // K=FF, N=E
  for (int d = 0; d < D_; ++d) {
    convT_kernel<<<tEE, 256, 0, stream>>>(Wq + (size_t)d * E_ * E_,  wtq + (size_t)d * E_ * E_,  E_,  E_);
    convT_kernel<<<tEE, 256, 0, stream>>>(Wk + (size_t)d * E_ * E_,  wtk + (size_t)d * E_ * E_,  E_,  E_);
    convT_kernel<<<tEE, 256, 0, stream>>>(Wv + (size_t)d * E_ * E_,  wtv + (size_t)d * E_ * E_,  E_,  E_);
    convT_kernel<<<tEE, 256, 0, stream>>>(Wo + (size_t)d * E_ * E_,  wto + (size_t)d * E_ * E_,  E_,  E_);
    convT_kernel<<<tEF, 256, 0, stream>>>(W1 + (size_t)d * E_ * FF_, wt1 + (size_t)d * E_ * FF_, E_,  FF_);
    convT_kernel<<<tFE, 256, 0, stream>>>(W2 + (size_t)d * FF_ * E_, wt2 + (size_t)d * FF_ * E_, FF_, E_);
  }

  const int ROWS = B_ * L_;                      // 4096
  dim3 gemmE(E_ / 64, ROWS / 256);               // (12, 16)
  dim3 gemmF(FF_ / 64, ROWS / 256);              // (48, 16)

  embed_kernel<<<ROWS, 256, 0, stream>>>(tokens, embed, e);
  dist_kernel<<<dim3(L_, B_), 128, 0, stream>>>(r, dmat);

  for (int d = 0; d < D_; ++d) {
    ln_kernel<<<ROWS, 256, 0, stream>>>(e, nullptr,
        ln1_g + (size_t)d * E_, ln1_b + (size_t)d * E_, e0);

    gemm_bias_kernel<<<gemmE, 256, 0, stream>>>(e0, wtq + (size_t)d * E_ * E_,
        bq + (size_t)d * E_, qb, ROWS, E_, E_, 0);
    gemm_bias_kernel<<<gemmE, 256, 0, stream>>>(e0, wtk + (size_t)d * E_ * E_,
        bk + (size_t)d * E_, kb, ROWS, E_, E_, 0);
    gemm_bias_kernel<<<gemmE, 256, 0, stream>>>(e0, wtv + (size_t)d * E_ * E_,
        bv + (size_t)d * E_, vb, ROWS, E_, E_, 0);

    attn_kernel<<<B_ * H_, 256, 0, stream>>>(qb, kb, vb, dmat, padding,
                                             mu, sigma, scalep, d, ob);

    gemm_bias_kernel<<<gemmE, 256, 0, stream>>>(ob, wto + (size_t)d * E_ * E_,
        bo + (size_t)d * E_, e1, ROWS, E_, E_, 0);

    ln_kernel<<<ROWS, 256, 0, stream>>>(e1, e0,
        ln2_g + (size_t)d * E_, ln2_b + (size_t)d * E_, e2);

    gemm_bias_kernel<<<gemmF, 256, 0, stream>>>(e2, wt1 + (size_t)d * E_ * FF_,
        b1 + (size_t)d * FF_, hbuf, ROWS, FF_, E_, 1);
    gemm_bias_kernel<<<gemmE, 256, 0, stream>>>(hbuf, wt2 + (size_t)d * FF_ * E_,
        b2 + (size_t)d * E_, e3, ROWS, E_, FF_, 0);

    addmask_kernel<<<(int)(SZ_E / 256), 256, 0, stream>>>(e3, e2, padding, e);
  }

  ln_kernel<<<ROWS, 256, 0, stream>>>(e, nullptr, lnf_g, lnf_b, e0);
  pool_kernel<<<dim3(E_ / 256, B_), 256, 0, stream>>>(e0, pooled);
  out_kernel<<<B_, 256, 0, stream>>>(pooled, Wout, bout, (float*)d_out);
}
